// CConvAggregationBlock_38981123179034
// MI455X (gfx1250) — compile-verified, matrix operand order checked by reading, not run on hardware
//
#include <hip/hip_runtime.h>

// Problem constants (match reference)
#define N_IN    100000
#define N_OUT   50000
#define KNBR    32
#define CIN     16
#define COUT    32
#define KS      4
#define NTAP    64                  // KS^3
#define KDIM    (NTAP * CIN)        // 1024 = GEMM K dimension
#define MTILE   16                  // outputs per block (WMMA M)
#define EDGES_PER_BLOCK (MTILE * KNBR)  // 512
#define THREADS 256

// Padded LDS layouts (bank-conflict avoidance; 64 banks x 4B)
#define BF_TAP_STRIDE 17            // 17 coprime with 64 -> taps spread over all banks
#define BF_ROW_STRIDE (NTAP * BF_TAP_STRIDE)   // 1088 floats per output row
#define WH_ROW_STRIDE (KDIM + 16)   // 1040 halves -> row stride 520 words (mod 64 = 8)

// LDS byte offsets
#define OFF_BF   0                                      // float [16][1088] = 69632 B
#define OFF_WH   (MTILE * BF_ROW_STRIDE * 4)            // 69632: f16 [32][1040] = 66560 B
#define OFF_DEN  (OFF_WH + COUT * WH_ROW_STRIDE * 2)    // 136192: float [16] = 64 B
#define LDS_BYTES (OFF_DEN + MTILE * 4)                 // 136256 -> 2 blocks / 320KB WGP

typedef __attribute__((ext_vector_type(16))) _Float16 v16h;
typedef __attribute__((ext_vector_type(8)))  float    v8f;

union Frag16 { v16h v; unsigned u[8]; };

extern "C" __global__ __launch_bounds__(THREADS)
void cconv_fused_wmma(const float* __restrict__ feats,        // [N_IN,16]
                      const float* __restrict__ inp_points,   // [N_IN,3]
                      const float* __restrict__ out_points,   // [N_OUT,3]
                      const float* __restrict__ out_extents,  // [N_OUT]
                      const float* __restrict__ scale_compat, // [E]
                      const int*   __restrict__ nbr_index,    // [E]
                      const float* __restrict__ nbr_dist,     // [E]
                      const float* __restrict__ kern,         // [64,16,32] row-major
                      const float* __restrict__ bias,         // [32]
                      float*       __restrict__ out)          // [N_OUT,32]
{
    extern __shared__ char smem[];
    float*    Bf  = (float*)   (smem + OFF_BF);   // fp32 tap accumulator (padded)
    _Float16* Wh  = (_Float16*)(smem + OFF_WH);   // f16 weights, [cout][kc] (padded)
    float*    den = (float*)   (smem + OFF_DEN);  // per-output importance sum -> recip

    const int tid = threadIdx.x;
    const int blk = blockIdx.x;

    // ---- phase 0: zero accumulators ------------------------------------
    for (int i = tid; i < MTILE * BF_ROW_STRIDE; i += THREADS) Bf[i] = 0.0f;
    if (tid < MTILE) den[tid] = 0.0f;
    __syncthreads();

    // ---- phase 1a: stage kernel weights into LDS as f16, transposed ----
    // kern flat index j = kc*32 + d  (kc = tap*16 + cin, d = cout)
    for (int j = tid; j < KDIM * COUT; j += THREADS) {
        const int kc = j >> 5;
        const int d  = j & 31;
        Wh[d * WH_ROW_STRIDE + kc] = (_Float16)kern[j];
    }

    // ---- phase 1b: edge scatter (512 edges, 2 per thread) --------------
    #pragma unroll
    for (int rep = 0; rep < 2; ++rep) {
        const int elocal = tid + rep * THREADS;            // 0..511
        const int e      = blk * EDGES_PER_BLOCK + elocal;
        const int olocal = elocal >> 5;                    // 0..15
        const int o      = blk * MTILE + olocal;

        const int   idx = nbr_index[e];
        const float dd  = nbr_dist[e];
        float w6 = 1.0f - dd * dd;
        w6 = w6 * w6 * w6;
        w6 = fminf(fmaxf(w6, 0.0f), 1.0f);
        const float imp = scale_compat[e] * w6;

        const float inv = 1.0f / (0.5f * out_extents[o]);
        const float rx = (inp_points[idx * 3 + 0] - out_points[o * 3 + 0]) * inv;
        const float ry = (inp_points[idx * 3 + 1] - out_points[o * 3 + 1]) * inv;
        const float rz = (inp_points[idx * 3 + 2] - out_points[o * 3 + 2]) * inv;

        // ball_to_cube_radial: u = rel * |rel|_2 / max(|rel|_inf, eps)
        const float rr = sqrtf(rx * rx + ry * ry + rz * rz);
        const float li = fmaxf(fmaxf(fabsf(rx), fabsf(ry)), fabsf(rz));
        const float s  = rr / fmaxf(li, 1e-12f);
        const float ux = rx * s, uy = ry * s, uz = rz * s;

        // trilinear coords in [0, KS-1]
        const float tx = fminf(fmaxf((ux * 0.5f + 0.5f) * (KS - 1), 0.0f), (float)(KS - 1));
        const float ty = fminf(fmaxf((uy * 0.5f + 0.5f) * (KS - 1), 0.0f), (float)(KS - 1));
        const float tz = fminf(fmaxf((uz * 0.5f + 0.5f) * (KS - 1), 0.0f), (float)(KS - 1));
        const float t0x = fminf(floorf(tx), (float)(KS - 2));
        const float t0y = fminf(floorf(ty), (float)(KS - 2));
        const float t0z = fminf(floorf(tz), (float)(KS - 2));
        const float fx = tx - t0x, fy = ty - t0y, fz = tz - t0z;
        const int ix0 = (int)t0x, iy0 = (int)t0y, iz0 = (int)t0z;

        // importance-weighted feature vector (vector loads; table lives in L2)
        const float4* f4 = (const float4*)(feats + (size_t)idx * CIN);
        const float4 a0 = f4[0], a1 = f4[1], a2 = f4[2], a3 = f4[3];
        float fe[16];
        fe[0]=a0.x*imp; fe[1]=a0.y*imp; fe[2]=a0.z*imp; fe[3]=a0.w*imp;
        fe[4]=a1.x*imp; fe[5]=a1.y*imp; fe[6]=a1.z*imp; fe[7]=a1.w*imp;
        fe[8]=a2.x*imp; fe[9]=a2.y*imp; fe[10]=a2.z*imp; fe[11]=a2.w*imp;
        fe[12]=a3.x*imp; fe[13]=a3.y*imp; fe[14]=a3.z*imp; fe[15]=a3.w*imp;

        atomicAdd(&den[olocal], imp);                      // ds_add_f32

        #pragma unroll
        for (int dx = 0; dx < 2; ++dx) {
            const float wx = dx ? fx : 1.0f - fx;
            const int   ix = ix0 + dx;
            #pragma unroll
            for (int dy = 0; dy < 2; ++dy) {
                const float wy = dy ? fy : 1.0f - fy;
                const int   iy = iy0 + dy;
                #pragma unroll
                for (int dz = 0; dz < 2; ++dz) {
                    const float wz = dz ? fz : 1.0f - fz;
                    const int   iz = iz0 + dz;
                    const float w = wx * wy * wz;
                    const int kflat = (ix * KS + iy) * KS + iz;
                    // padded stride 17: distinct taps -> distinct LDS banks
                    float* dst = Bf + olocal * BF_ROW_STRIDE + kflat * BF_TAP_STRIDE;
                    #pragma unroll
                    for (int c = 0; c < CIN; ++c)
                        atomicAdd(dst + c, fe[c] * w);     // ds_add_f32 (LDS)
                }
            }
        }
    }
    __syncthreads();

    // ---- phase 2: denominator -> reciprocal (kills per-lane divides) ----
    if (tid < MTILE) {
        const float dn = den[tid];
        den[tid] = (dn > 0.0f) ? (1.0f / dn) : 1.0f;
    }
    __syncthreads();

    // ---- phase 3: [16 x 1024] x [1024 x 32] via v_wmma_f32_16x16x32_f16
    // wave 0 -> cout 0..15, wave 1 -> cout 16..31; f32->f16 done in-register
    const int wave = tid >> 5;
    if (wave < 2) {
        const int lane = tid & 31;
        const int col  = lane & 15;    // A row m / B col n / C col n
        const int grp  = lane >> 4;    // K-octet group
        const int n0   = wave * 16;

        // CDNA5 16-bit A/B layout: lane-group g holds K octets {8g, 8g+16..}
        int kb[8];
        #pragma unroll
        for (int j = 0; j < 8; ++j)
            kb[j] = (j < 4 ? 2 * j : 2 * (j - 4) + 16) + 8 * grp;

        const float*    arow = Bf + col * BF_ROW_STRIDE;
        const _Float16* brow = Wh + (n0 + col) * WH_ROW_STRIDE;

        v8f acc = {};
        for (int kc = 0; kc < KDIM / 32; ++kc) {           // 32 K-steps
            Frag16 a, b;
            #pragma unroll
            for (int j = 0; j < 8; ++j) {
                const int K0  = kc * 32 + kb[j];           // even
                const int tap = K0 >> 4;
                const int ch  = K0 & 15;                   // even, ch+1 same tap
                const float f0 = arow[tap * BF_TAP_STRIDE + ch];
                const float f1 = arow[tap * BF_TAP_STRIDE + ch + 1];
                a.v[2 * j]     = (_Float16)f0;
                a.v[2 * j + 1] = (_Float16)f1;
                __builtin_memcpy(&b.u[j], brow + K0, 4);
            }
            acc = __builtin_amdgcn_wmma_f32_16x16x32_f16(
                      /*neg_a=*/false, a.v, /*neg_b=*/false, b.v,
                      /*c_mod=*/(short)0, acc,
                      /*reuse_a=*/false, /*reuse_b=*/false);
        }

        // epilogue: C layout -> VGPR r, lanes 0-15 M=r, lanes 16-31 M=r+8
        const float bv = bias[n0 + col];
        #pragma unroll
        for (int r = 0; r < 8; ++r) {
            const int mrow = r + 8 * grp;
            const float dninv = den[mrow];                 // reciprocal (phase 2)
            float v = acc[r] * dninv + bv;
            v = fmaxf(v, 0.0f);
            out[((size_t)(blk * MTILE + mrow)) * COUT + n0 + col] = v;
        }
    }
}

extern "C" void kernel_launch(void* const* d_in, const int* in_sizes, int n_in,
                              void* d_out, int out_size, void* d_ws, size_t ws_size,
                              hipStream_t stream) {
    const float* feats        = (const float*)d_in[0];
    const float* inp_points   = (const float*)d_in[1];
    const float* out_points   = (const float*)d_in[2];
    const float* out_extents  = (const float*)d_in[3];
    const float* scale_compat = (const float*)d_in[4];
    const int*   nbr_index    = (const int*)  d_in[5];
    // d_in[6] = neighbors_row_splits: uniform K=32, out_id = e >> 5 (unused)
    const float* nbr_dist     = (const float*)d_in[7];
    const float* kern         = (const float*)d_in[8];
    const float* bias         = (const float*)d_in[9];
    float* out = (float*)d_out;

    const int nblocks = N_OUT / MTILE;  // 3125 (exact)
    cconv_fused_wmma<<<dim3(nblocks), dim3(THREADS), LDS_BYTES, stream>>>(
        feats, inp_points, out_points, out_extents, scale_compat,
        nbr_index, nbr_dist, kern, bias, out);
}
